// GPTNeoGQASelfAttention_56289841381522
// MI455X (gfx1250) — compile-verified, matrix operand order checked by reading, not run on hardware
//
#include <hip/hip_runtime.h>
#include <hip/hip_bf16.h>

typedef __bf16 bf16_t;
typedef __attribute__((ext_vector_type(16))) __bf16 bf16x16;
typedef __attribute__((ext_vector_type(8)))  __bf16 bf16x8;
typedef __attribute__((ext_vector_type(8)))  float  f32x8;
typedef __attribute__((ext_vector_type(4)))  float  f32x4;

#define DEVINL __device__ __forceinline__

// Load a 16-element bf16 A/B fragment for v_wmma_f32_16x16x32_bf16.
// Per ISA layout: lanes 0-15 hold K = {0..7, 16..23}, lanes 16-31 hold
// K = {8..15, 24..31}.  rowp points at K=0 of this lane's row; half8 = (lane>=16)?8:0.
DEVINL bf16x16 load_frag(const bf16_t* rowp, int half8) {
  bf16x8 lo = *(const bf16x8*)(rowp + half8);
  bf16x8 hi = *(const bf16x8*)(rowp + 16 + half8);
  return __builtin_shufflevector(lo, hi, 0,1,2,3,4,5,6,7,8,9,10,11,12,13,14,15);
}

DEVINL f32x8 wmma_bf16(bf16x16 a, bf16x16 b, f32x8 c) {
  return __builtin_amdgcn_wmma_f32_16x16x32_bf16(false, a, false, b, (short)0, c,
                                                 false, false);
}

// CDNA5 async DMA: copy 16B per lane from global directly into LDS, tracked
// with ASYNCcnt (no VGPR round trip).  GV mode: 64-bit vaddr, LDS offset vdst.
DEVINL void async_load_b128(unsigned lds_off, const void* gaddr) {
  asm volatile("global_load_async_to_lds_b128 %0, %1, off"
               :: "v"(lds_off), "v"(gaddr)
               : "memory");
}
DEVINL void wait_async() {
  asm volatile("s_wait_asynccnt 0x0" ::: "memory");
}
// Flat shared-aperture addresses carry the workgroup-relative LDS offset in
// the low 32 bits (ISA 10.2).
DEVINL unsigned lds_off_of(const void* p) {
  return (unsigned)(unsigned long long)p;
}

// ---------------------------------------------------------------------------
// GEMM: Out[M,N] = A[M,K] * W[K,N] (+ bias), A fp32 or bf16, W fp32.
// 128x128 block, BK=32, 256 threads = 8 waves; wave = 32x64 sub-tile (2x4 WMMA).
// ---------------------------------------------------------------------------
template <bool A_BF16, bool OUT_F32_BIAS>
__global__ void __launch_bounds__(256)
gemm_kernel(const void* __restrict__ Ain, const float* __restrict__ W,
            const float* __restrict__ bias, void* __restrict__ Out,
            int M, int N, int K) {
  constexpr int BM = 128, BN = 128, BK = 32;
  constexpr int AST = BK + 16;  // 48 bf16 = 96B row stride (16B aligned)
  __shared__ bf16_t As[BM * AST];
  __shared__ bf16_t Bs[BN * AST];  // W tile stored transposed: [n][k]

  const int tid  = threadIdx.x;
  const int lane = tid & 31;
  const int wv   = tid >> 5;
  const int half8 = (lane >> 4) * 8;
  const int l16   = lane & 15;
  const int wm = wv & 3;   // wave row group: rows wm*32
  const int wn = wv >> 2;  // wave col group: cols wn*64

  const int n0 = blockIdx.x * BN;
  const int m0 = blockIdx.y * BM;

  const f32x8 zerov = {0.f, 0.f, 0.f, 0.f, 0.f, 0.f, 0.f, 0.f};
  f32x8 acc[2][4];
#pragma unroll
  for (int i = 0; i < 2; i++)
#pragma unroll
    for (int j = 0; j < 4; j++) acc[i][j] = zerov;

  const int srow  = tid >> 1;  // 0..127
  const int shalf = tid & 1;   // 0/1 -> 16-element half of BK

  for (int k0 = 0; k0 < K; k0 += BK) {
    __syncthreads();
    // ---- stage A tile (convert fp32 -> bf16 if needed) ----
    if constexpr (A_BF16) {
      const bf16_t* src = (const bf16_t*)Ain + (size_t)(m0 + srow) * K + k0 + shalf * 16;
      *(bf16x8*)&As[srow * AST + shalf * 16]     = *(const bf16x8*)(src);
      *(bf16x8*)&As[srow * AST + shalf * 16 + 8] = *(const bf16x8*)(src + 8);
      if (k0 + BK < K) __builtin_prefetch(src + BK, 0, 0);
    } else {
      const float* src = (const float*)Ain + (size_t)(m0 + srow) * K + k0 + shalf * 16;
      f32x4 f0 = *(const f32x4*)(src);
      f32x4 f1 = *(const f32x4*)(src + 4);
      f32x4 f2 = *(const f32x4*)(src + 8);
      f32x4 f3 = *(const f32x4*)(src + 12);
      bf16x8 v0, v1;
#pragma unroll
      for (int j = 0; j < 4; j++) {
        v0[j]     = (bf16_t)f0[j];
        v0[4 + j] = (bf16_t)f1[j];
        v1[j]     = (bf16_t)f2[j];
        v1[4 + j] = (bf16_t)f3[j];
      }
      *(bf16x8*)&As[srow * AST + shalf * 16]     = v0;
      *(bf16x8*)&As[srow * AST + shalf * 16 + 8] = v1;
      if (k0 + BK < K) __builtin_prefetch(src + BK, 0, 0);
    }
    // ---- stage W tile transposed: Bs[n][k] = W[k0+k][n0+n] ----
    {
      const int bk   = tid >> 3;  // 0..31
      const int nseg = tid & 7;   // n base = nseg*16
      const float* src = W + (size_t)(k0 + bk) * N + n0 + nseg * 16;
      f32x4 f0 = *(const f32x4*)(src);
      f32x4 f1 = *(const f32x4*)(src + 4);
      f32x4 f2 = *(const f32x4*)(src + 8);
      f32x4 f3 = *(const f32x4*)(src + 12);
#pragma unroll
      for (int j = 0; j < 4; j++) {
        Bs[(nseg * 16 + j) * AST + bk]      = (bf16_t)f0[j];
        Bs[(nseg * 16 + 4 + j) * AST + bk]  = (bf16_t)f1[j];
        Bs[(nseg * 16 + 8 + j) * AST + bk]  = (bf16_t)f2[j];
        Bs[(nseg * 16 + 12 + j) * AST + bk] = (bf16_t)f3[j];
      }
      if (k0 + BK < K) __builtin_prefetch(src + (size_t)BK * N, 0, 0);
    }
    __syncthreads();

    bf16x16 af[2], bfr[4];
#pragma unroll
    for (int mt = 0; mt < 2; mt++)
      af[mt] = load_frag(&As[(wm * 32 + mt * 16 + l16) * AST], half8);
#pragma unroll
    for (int nt = 0; nt < 4; nt++)
      bfr[nt] = load_frag(&Bs[(wn * 64 + nt * 16 + l16) * AST], half8);
#pragma unroll
    for (int mt = 0; mt < 2; mt++)
#pragma unroll
      for (int nt = 0; nt < 4; nt++)
        acc[mt][nt] = wmma_bf16(af[mt], bfr[nt], acc[mt][nt]);
  }

  // ---- epilogue: C/D layout -> global ----
#pragma unroll
  for (int mt = 0; mt < 2; mt++) {
#pragma unroll
    for (int nt = 0; nt < 4; nt++) {
      const int gm = m0 + wm * 32 + mt * 16 + half8;
      const int gn = n0 + wn * 64 + nt * 16 + l16;
      if constexpr (OUT_F32_BIAS) {
        const float bv = bias[gn];
        float* Of = (float*)Out;
#pragma unroll
        for (int r = 0; r < 8; r++)
          Of[(size_t)(gm + r) * N + gn] = acc[mt][nt][r] + bv;
      } else {
        bf16_t* Ob = (bf16_t*)Out;
#pragma unroll
        for (int r = 0; r < 8; r++)
          Ob[(size_t)(gm + r) * N + gn] = (bf16_t)acc[mt][nt][r];
      }
    }
  }
}

// ---------------------------------------------------------------------------
// Flash-style causal GQA attention (GPT-Neo: NO 1/sqrt(d) scaling).
// One workgroup per (128-query tile, head, batch); 8 waves, 16 q-rows/wave.
// Streams 128-key blocks; online softmax in fp32; bf16 WMMA for QK^T and P@V.
// Q/K tiles staged with CDNA5 async global->LDS DMA (ASYNCcnt).
// ---------------------------------------------------------------------------
__global__ void __launch_bounds__(256)
attn_kernel(const bf16_t* __restrict__ Q, const bf16_t* __restrict__ Kb,
            const bf16_t* __restrict__ Vb, bf16_t* __restrict__ Ctx) {
  constexpr int S = 2048, HD = 128, NHD = 2048, KVD = 512;
  constexpr int STR = 144;  // padded row stride (16B-aligned, conflict-reducing)
  extern __shared__ bf16_t smem[];
  bf16_t* Qs = smem;                  // [128][STR] queries (row-major over hd)
  bf16_t* Ks = smem + 128 * STR;      // [128][STR] keys   (row-major over hd)
  bf16_t* Vt = smem + 2 * 128 * STR;  // [HD][STR]  V transposed: [hd][key]
  bf16_t* Ps = smem + 3 * 128 * STR;  // [8 waves][16][STR] probabilities

  const int qb = blockIdx.x;  // query tile 0..15
  const int h  = blockIdx.y;  // head 0..15
  const int b  = blockIdx.z;  // batch 0..3
  const int kv = h >> 2;      // GQA: 4 consecutive q-heads per kv head

  const int tid  = threadIdx.x;
  const int lane = tid & 31;
  const int wv   = tid >> 5;
  const int half8 = (lane >> 4) * 8;
  const int l16   = lane & 15;

  const unsigned qs_off = lds_off_of(Qs);
  const unsigned ks_off = lds_off_of(Ks);

  // stage Q tile once via async DMA: rows qb*128+r, cols h*HD..h*HD+127
#pragma unroll
  for (int i = 0; i < 8; i++) {
    const int idx = tid + i * 256;
    const int r = idx >> 4, seg = idx & 15;
    const bf16_t* src = Q + (size_t)(b * S + qb * 128 + r) * NHD + h * HD + seg * 8;
    async_load_b128(qs_off + (unsigned)(r * STR + seg * 8) * 2, src);
  }

  const f32x8 zerov = {0.f, 0.f, 0.f, 0.f, 0.f, 0.f, 0.f, 0.f};
  float mrow[8], lrow[8];
  f32x8 o[8];
#pragma unroll
  for (int r = 0; r < 8; r++) { mrow[r] = -3.0e38f; lrow[r] = 0.f; }
#pragma unroll
  for (int nt = 0; nt < 8; nt++) o[nt] = zerov;

  for (int kb = 0; kb <= qb; kb++) {
    __syncthreads();  // previous block's compute done before restaging
    // stage K block via async DMA
#pragma unroll
    for (int i = 0; i < 8; i++) {
      const int idx = tid + i * 256;
      const int r = idx >> 4, seg = idx & 15;
      const bf16_t* src = Kb + (size_t)(b * S + kb * 128 + r) * KVD + kv * HD + seg * 8;
      async_load_b128(ks_off + (unsigned)(r * STR + seg * 8) * 2, src);
    }
    // stage V block transposed: Vt[hd][key] (manual: DMA cannot transpose)
#pragma unroll
    for (int i = 0; i < 8; i++) {
      const int idx = tid + i * 256;
      const int key = idx >> 4, seg = idx & 15, d0 = seg * 8;
      bf16x8 v = *(const bf16x8*)(Vb + (size_t)(b * S + kb * 128 + key) * KVD + kv * HD + d0);
#pragma unroll
      for (int j = 0; j < 8; j++) Vt[(d0 + j) * STR + key] = v[j];
    }
    wait_async();     // this wave's async Q/K tiles landed in LDS
    __syncthreads();  // all waves' staging visible

    // ---- scores: S = Q_tile(16xHD) @ K_block^T  (per wave) ----
    bf16x16 qf[4];
#pragma unroll
    for (int ks = 0; ks < 4; ks++)
      qf[ks] = load_frag(&Qs[(wv * 16 + l16) * STR + ks * 32], half8);
    f32x8 sc[8];
#pragma unroll
    for (int nt = 0; nt < 8; nt++) {
      f32x8 a = zerov;
#pragma unroll
      for (int ks = 0; ks < 4; ks++) {
        bf16x16 kf = load_frag(&Ks[(nt * 16 + l16) * STR + ks * 32], half8);
        a = wmma_bf16(qf[ks], kf, a);
      }
      sc[nt] = a;
    }

    // causal mask on the diagonal block
    if (kb == qb) {
#pragma unroll
      for (int nt = 0; nt < 8; nt++) {
        const int key = nt * 16 + l16;
#pragma unroll
        for (int r = 0; r < 8; r++) {
          const int qr = wv * 16 + half8 + r;
          if (key > qr) sc[nt][r] = -3.0e38f;
        }
      }
    }

    // ---- online softmax (per q row; rows live across 16 lanes of a half) ----
#pragma unroll
    for (int r = 0; r < 8; r++) {
      float mx = -3.0e38f;
#pragma unroll
      for (int nt = 0; nt < 8; nt++) mx = fmaxf(mx, sc[nt][r]);
#pragma unroll
      for (int off = 1; off < 16; off <<= 1) mx = fmaxf(mx, __shfl_xor(mx, off, 32));
      const float mnew = fmaxf(mrow[r], mx);
      const float corr = __expf(mrow[r] - mnew);
      float rs = 0.f;
#pragma unroll
      for (int nt = 0; nt < 8; nt++) {
        const float p = __expf(sc[nt][r] - mnew);
        sc[nt][r] = p;
        rs += p;
      }
#pragma unroll
      for (int off = 1; off < 16; off <<= 1) rs += __shfl_xor(rs, off, 32);
      mrow[r] = mnew;
      lrow[r] = lrow[r] * corr + rs;
#pragma unroll
      for (int nt = 0; nt < 8; nt++) o[nt][r] *= corr;
    }

    // ---- P (C-layout regs) -> per-wave LDS -> A-layout fragments ----
    bf16_t* pw = Ps + wv * 16 * STR;
#pragma unroll
    for (int nt = 0; nt < 8; nt++)
#pragma unroll
      for (int r = 0; r < 8; r++)
        pw[(half8 + r) * STR + nt * 16 + l16] = (bf16_t)sc[nt][r];

    // ---- ctx += P(16x128) @ V(128xHD) ----
#pragma unroll
    for (int ks = 0; ks < 4; ks++) {
      bf16x16 pf = load_frag(pw + l16 * STR + ks * 32, half8);
#pragma unroll
      for (int nt = 0; nt < 8; nt++) {
        bf16x16 vf = load_frag(&Vt[(nt * 16 + l16) * STR + ks * 32], half8);
        o[nt] = wmma_bf16(pf, vf, o[nt]);
      }
    }
  }

  // ---- epilogue: ctx /= l, write bf16 [B*S, NH*HD] ----
#pragma unroll
  for (int r = 0; r < 8; r++) lrow[r] = 1.f / lrow[r];
#pragma unroll
  for (int nt = 0; nt < 8; nt++) {
#pragma unroll
    for (int r = 0; r < 8; r++) {
      const size_t row = (size_t)(b * S + qb * 128 + wv * 16 + half8 + r);
      Ctx[row * NHD + h * HD + nt * 16 + l16] = (bf16_t)(o[nt][r] * lrow[r]);
    }
  }
}

// ---------------------------------------------------------------------------
extern "C" void kernel_launch(void* const* d_in, const int* in_sizes, int n_in,
                              void* d_out, int out_size, void* d_ws, size_t ws_size,
                              hipStream_t stream) {
  (void)in_sizes; (void)n_in; (void)out_size; (void)ws_size;
  const float* X  = (const float*)d_in[0];
  const float* qw = (const float*)d_in[1];
  const float* kw = (const float*)d_in[2];
  const float* vw = (const float*)d_in[3];
  const float* ow = (const float*)d_in[4];
  const float* ob = (const float*)d_in[5];
  float* out = (float*)d_out;

  // workspace: Qb 32MB | Kb 8MB | Vb 8MB | Ctx 32MB  (bf16), total ~80MB
  char* ws = (char*)d_ws;
  const size_t QB = (size_t)8192 * 2048 * 2;
  const size_t KB = (size_t)8192 * 512 * 2;
  bf16_t* Qb  = (bf16_t*)(ws);
  bf16_t* Kbf = (bf16_t*)(ws + QB);
  bf16_t* Vbf = (bf16_t*)(ws + QB + KB);
  bf16_t* Ctx = (bf16_t*)(ws + QB + 2 * KB);

  dim3 blk(256);
  // Q/K/V projections (fp32 in, bf16 out)
  gemm_kernel<false, false><<<dim3(16, 64), blk, 0, stream>>>(X, qw, nullptr, Qb, 8192, 2048, 2048);
  gemm_kernel<false, false><<<dim3(4, 64),  blk, 0, stream>>>(X, kw, nullptr, Kbf, 8192, 512, 2048);
  gemm_kernel<false, false><<<dim3(4, 64),  blk, 0, stream>>>(X, vw, nullptr, Vbf, 8192, 512, 2048);

  // attention: 16 q-tiles x 16 heads x 4 batches, 144KB dynamic LDS
  const size_t smem_bytes = (size_t)4 * 128 * 144 * sizeof(bf16_t);  // 147456
  attn_kernel<<<dim3(16, 16, 4), blk, smem_bytes, stream>>>(Qb, Kbf, Vbf, Ctx);

  // out projection (bf16 in, fp32 out, + bias)
  gemm_kernel<true, true><<<dim3(16, 64), blk, 0, stream>>>(Ctx, ow, ob, out, 8192, 2048, 2048);
}
